// MultiHeadAttention2d_23184233464283
// MI455X (gfx1250) — matrix operand order repaired and checked by reading, not compiled, not run on hardware
//
#include <hip/hip_runtime.h>
#include <hip/hip_bf16.h>

// ---------------------------------------------------------------------------
// Types for CDNA5 WMMA (wave32, 16x16x32 bf16 -> f32)
// ---------------------------------------------------------------------------
typedef unsigned short u16;
typedef u16    u16x8  __attribute__((ext_vector_type(8)));
typedef u16    u16x16 __attribute__((ext_vector_type(16)));
typedef __bf16 bf16x16 __attribute__((ext_vector_type(16)));
typedef float  f32x8  __attribute__((ext_vector_type(8)));
typedef int    i32x4  __attribute__((ext_vector_type(4)));

// gfx1250 async global->LDS copy path (ASYNCcnt-tracked), guarded so the file
// compiles on toolchains where the builtins are absent. The builtin takes
// (int4 AS1* src, int4 AS3* dst, imm offset, imm cpol).
#if __has_builtin(__builtin_amdgcn_global_load_async_to_lds_b128) && \
    __has_builtin(__builtin_amdgcn_s_wait_asynccnt)
#define ATTN_ASYNC 1
#define AS1 __attribute__((address_space(1)))
#define AS3 __attribute__((address_space(3)))
#endif

static __device__ __forceinline__ f32x8 wmma_bf16(bf16x16 a, bf16x16 b, f32x8 c) {
  // (neg_a, A, neg_b, B, c_mod, C, reuse_a, reuse_b)
  return __builtin_amdgcn_wmma_f32_16x16x32_bf16(false, a, false, b, (short)0, c,
                                                 false, false);
}

// float -> bf16 bits, round-to-nearest-even
static __device__ __forceinline__ u16 f2bf(float f) {
  unsigned int u = __float_as_uint(f);
  unsigned int r = u + 0x7FFFu + ((u >> 16) & 1u);
  return (u16)(r >> 16);
}

// Build a 16-element bf16 operand from two 16B chunks (A-operand K-interleave:
// lane<16 holds K {0..7, 16..23}+win, lane>=16 holds K {8..15, 24..31}+win).
static __device__ __forceinline__ bf16x16 mk2(const u16* p0, const u16* p1) {
  u16x8 lo = *(const u16x8*)p0;
  u16x8 hi = *(const u16x8*)p1;
  u16x16 v = __builtin_shufflevector(lo, hi, 0, 1, 2, 3, 4, 5, 6, 7,
                                     8, 9, 10, 11, 12, 13, 14, 15);
  return __builtin_bit_cast(bf16x16, v);
}

// ---------------------------------------------------------------------------
// Problem constants: B=8, C=512, H=W=32 -> N=1024, heads=8, dim=64, 3C=1536
// ---------------------------------------------------------------------------
#define CB   8
#define CC   512
#define CN   1024
#define CH   8
#define CD   64
#define CO3  1536

// LDS row pitches (halfwords) chosen to spread banks / keep 16B alignment
#define KPITCH 72
#define VPITCH 40
#define PPITCH 40

// ---------------------------------------------------------------------------
// Kernel 1: weight norm  w[o][c] = g[o] * v[o][c] * rsqrt(sum_c v^2) -> bf16
// ---------------------------------------------------------------------------
__global__ __launch_bounds__(256) void k_wnorm(const float* __restrict__ pv,
                                               const float* __restrict__ pg,
                                               u16* __restrict__ wbf) {
  __shared__ float red[256];
  const int o = blockIdx.x;
  const int t = threadIdx.x;
  const float* row = pv + (size_t)o * CC;
  float a = row[t];
  float b = row[t + 256];
  red[t] = a * a + b * b;
  __syncthreads();
  for (int s = 128; s > 0; s >>= 1) {
    if (t < s) red[t] += red[t + s];
    __syncthreads();
  }
  float scale = pg[o] * rsqrtf(red[0]);
  wbf[(size_t)o * CC + t]       = f2bf(a * scale);
  wbf[(size_t)o * CC + t + 256] = f2bf(b * scale);
}

// ---------------------------------------------------------------------------
// Kernel 2: x (B,C,N) f32 -> xT (B,N,C) bf16  (so GEMM B columns are contiguous)
// ---------------------------------------------------------------------------
__global__ __launch_bounds__(256) void k_transpose_x(const float* __restrict__ x,
                                                     u16* __restrict__ xT) {
  const size_t t = (size_t)blockIdx.x * blockDim.x + threadIdx.x;  // (b,n,c)
  const int c = (int)(t & (CC - 1));
  const int n = (int)((t >> 9) & (CN - 1));
  const int b = (int)(t >> 19);
  xT[t] = f2bf(x[((size_t)b * CC + c) * CN + n]);
}

// ---------------------------------------------------------------------------
// Kernel 3: QKV GEMM.  c[b][o][n] = sum_c w[o][c]*x[b][c][n] + bias[o]
// Wave tile 32x32 (2x2 WMMA accum), WG = 8 waves = 64(M) x 128(N).
// Scatter: q -> qT[b][h][n][d] (scaled by 0.125), k -> kT[b][h][n][d],
//          v -> v[b][h][d][n].   All bf16.
// ---------------------------------------------------------------------------
__global__ __launch_bounds__(256) void k_qkv_gemm(const u16* __restrict__ wbf,
                                                  const u16* __restrict__ xT,
                                                  const float* __restrict__ bias,
                                                  u16* __restrict__ qT,
                                                  u16* __restrict__ kT,
                                                  u16* __restrict__ vv) {
  const int tid = threadIdx.x;
  const int l   = tid & 31;
  const int wid = tid >> 5;
  const int wm  = wid & 1;        // 2 waves in M
  const int wn  = wid >> 1;       // 4 waves in N
  const int lh  = l >> 4;         // lane half
  const int ll  = l & 15;

  const int ob = blockIdx.x * 64 + wm * 32;
  const int nb = blockIdx.y * 128 + wn * 32;
  const int b  = blockIdx.z;

  const u16* arow0 = wbf + (size_t)(ob + ll) * CC;
  const u16* arow1 = wbf + (size_t)(ob + 16 + ll) * CC;
  const u16* bcol0 = xT + ((size_t)b * CN + nb + ll) * CC;
  const u16* bcol1 = xT + ((size_t)b * CN + nb + 16 + ll) * CC;
  const int koff = lh * 8;   // A operand K-interleave offset
  const int boff = lh * 8;   // B operand K-interleave offset

  f32x8 acc00 = {}, acc01 = {}, acc10 = {}, acc11 = {};

  for (int c0 = 0; c0 < CC; c0 += 32) {
    bf16x16 a0 = mk2(arow0 + c0 + koff, arow0 + c0 + 16 + koff);
    bf16x16 a1 = mk2(arow1 + c0 + koff, arow1 + c0 + 16 + koff);
    bf16x16 b0 = mk2(bcol0 + c0 + boff, bcol0 + c0 + boff + 16);
    bf16x16 b1 = mk2(bcol1 + c0 + boff, bcol1 + c0 + boff + 16);
    acc00 = wmma_bf16(a0, b0, acc00);
    acc01 = wmma_bf16(a0, b1, acc01);
    acc10 = wmma_bf16(a1, b0, acc10);
    acc11 = wmma_bf16(a1, b1, acc11);
  }

  // Epilogue: C layout -> VGPR i, lane half: M = i + 8*lh, N = ll
  f32x8 acc[2][2] = {{acc00, acc01}, {acc10, acc11}};
#pragma unroll
  for (int mi = 0; mi < 2; ++mi) {
#pragma unroll
    for (int ni = 0; ni < 2; ++ni) {
#pragma unroll
      for (int i = 0; i < 8; ++i) {
        const int o = ob + mi * 16 + i + 8 * lh;
        const int n = nb + ni * 16 + ll;
        float val = acc[mi][ni][i] + bias[o];
        if (o < CC) {                       // Q (pre-scaled by 1/sqrt(dim))
          int h = o >> 6, d = o & 63;
          qT[((size_t)(b * CH + h) * CN + n) * CD + d] = f2bf(val * 0.125f);
        } else if (o < 2 * CC) {            // K
          int oo = o - CC, h = oo >> 6, d = oo & 63;
          kT[((size_t)(b * CH + h) * CN + n) * CD + d] = f2bf(val);
        } else {                            // V
          int oo = o - 2 * CC, h = oo >> 6, d = oo & 63;
          vv[((size_t)(b * CH + h) * CD + d) * CN + n] = f2bf(val);
        }
      }
    }
  }
}

// ---------------------------------------------------------------------------
// Kernel 4: flash attention + residual, with workgroup-shared K/V staging.
// Grid: (q-blocks=8, heads=8, batch=8); 8 waves/WG, 16 queries per wave.
// All 8 waves share (b,h), so each 32-row K tile (4KB) and 64x32 V tile (4KB)
// is DMA'd into LDS ONCE per workgroup (async-to-LDS, double-buffered) instead
// of 8x redundant global reads. Per wave: S^T = K.Q^T (M=k, N=q) so softmax
// over k is per-lane + xor16; P round-trips a wave-private LDS tile
// (C-layout -> B-operand); O += V.P^T with 4 WMMAs per block.
// ---------------------------------------------------------------------------
__global__ __launch_bounds__(256) void k_attn(const u16* __restrict__ qT,
                                              const u16* __restrict__ kT,
                                              const u16* __restrict__ vv,
                                              const float* __restrict__ x,
                                              float* __restrict__ out) {
  __shared__ u16 pLds[8 * 16 * PPITCH];     // per-wave P^T staging
  __shared__ u16 Ksm[2][32 * KPITCH];       // double-buffered K block (32 x 64)
  __shared__ u16 Vsm[2][64 * VPITCH];       // double-buffered V block (64 x 32)

  const int tid = threadIdx.x;
  const int l   = tid & 31;
  const int wid = tid >> 5;
  const int lh  = l >> 4;
  const int ll  = l & 15;

  const int b  = blockIdx.z;
  const int h  = blockIdx.y;
  const int bh = b * CH + h;
  const int q0 = blockIdx.x * 128 + wid * 16;

  const u16* kbase = kT + (size_t)bh * CN * CD;
  const u16* vbase = vv + (size_t)bh * CD * CN;

  // Cooperative staging map: 256 threads x one 16B chunk each per tile.
  const int rK = tid >> 3, cK = tid & 7;    // K: row 0..31, chunk 0..7
  const int rV = tid >> 2, cV = tid & 3;    // V: row 0..63, chunk 0..3

  auto stage = [&](int kb, int buf) {
    const u16* kg = kbase + (size_t)(kb + rK) * CD + cK * 8;
    u16* ks = &Ksm[buf][rK * KPITCH + cK * 8];
    const u16* vg = vbase + (size_t)rV * CN + kb + cV * 8;
    u16* vs = &Vsm[buf][rV * VPITCH + cV * 8];
#if defined(ATTN_ASYNC)
    __builtin_amdgcn_global_load_async_to_lds_b128((AS1 i32x4*)(u16*)kg,
                                                   (AS3 i32x4*)ks, 0, 0);
    __builtin_amdgcn_global_load_async_to_lds_b128((AS1 i32x4*)(u16*)vg,
                                                   (AS3 i32x4*)vs, 0, 0);
#else
    *(u16x8*)ks = *(const u16x8*)kg;
    *(u16x8*)vs = *(const u16x8*)vg;
#endif
  };

  // Fixed B-operand: Q columns (col q = ll), d split 0-31 / 32-63
  const u16* qrow = qT + ((size_t)bh * CN + q0 + ll) * CD;
  const bf16x16 bq0 = mk2(qrow + lh * 8, qrow + 16 + lh * 8);
  const bf16x16 bq1 = mk2(qrow + 32 + lh * 8, qrow + 48 + lh * 8);

  u16* pbase = pLds + (wid * 16 + ll) * PPITCH;  // this lane's q row in LDS

  f32x8 oacc[4] = {};
  float m_run = -INFINITY;
  float l_run = 0.0f;

  stage(0, 0);  // prologue: DMA block 0

  const int nblk = CN / 32;
  for (int ib = 0; ib < nblk; ++ib) {
    const int cur = ib & 1;
    if (ib + 1 < nblk) stage((ib + 1) * 32, cur ^ 1);  // overlap next DMA
#if defined(ATTN_ASYNC)
    if (ib + 1 < nblk) __builtin_amdgcn_s_wait_asynccnt(2);  // keep next in flight
    else               __builtin_amdgcn_s_wait_asynccnt(0);
#endif
    __syncthreads();  // block `ib` visible to all waves

    const u16* Ks = &Ksm[cur][0];
    const u16* Vs = &Vsm[cur][0];

    // ---- S^T block: two 16(k) x 16(q) tiles, K-dim = d = 64 (2 steps) ----
    const u16* k0p = Ks + (size_t)ll * KPITCH;
    const u16* k1p = Ks + (size_t)(16 + ll) * KPITCH;
    bf16x16 ak00 = mk2(k0p + lh * 8,      k0p + 16 + lh * 8);
    bf16x16 ak01 = mk2(k0p + 32 + lh * 8, k0p + 48 + lh * 8);
    bf16x16 ak10 = mk2(k1p + lh * 8,      k1p + 16 + lh * 8);
    bf16x16 ak11 = mk2(k1p + 32 + lh * 8, k1p + 48 + lh * 8);

    f32x8 z = {};
    f32x8 s0 = wmma_bf16(ak00, bq0, z);
    s0 = wmma_bf16(ak01, bq1, s0);
    f32x8 s1 = wmma_bf16(ak10, bq0, z);
    s1 = wmma_bf16(ak11, bq1, s1);

    // ---- online softmax over this 32-k block (per q = ll) ----
    float mb = s0[0];
#pragma unroll
    for (int i = 1; i < 8; ++i) mb = fmaxf(mb, s0[i]);
#pragma unroll
    for (int i = 0; i < 8; ++i) mb = fmaxf(mb, s1[i]);
    mb = fmaxf(mb, __shfl_xor(mb, 16, 32));

    const float m_new = fmaxf(m_run, mb);
    const float corr  = __expf(m_run - m_new);

    float e0[8], e1[8], sum = 0.0f;
#pragma unroll
    for (int i = 0; i < 8; ++i) { e0[i] = __expf(s0[i] - m_new); sum += e0[i]; }
#pragma unroll
    for (int i = 0; i < 8; ++i) { e1[i] = __expf(s1[i] - m_new); sum += e1[i]; }
    sum += __shfl_xor(sum, 16, 32);

    l_run = l_run * corr + sum;
    m_run = m_new;
#pragma unroll
    for (int t = 0; t < 4; ++t) oacc[t] = oacc[t] * corr;

    // ---- pack P^T to LDS: row q = ll, k local = i + 8*lh (+16 for tile1) ----
    u16x8 p0, p1;
#pragma unroll
    for (int i = 0; i < 8; ++i) { p0[i] = f2bf(e0[i]); p1[i] = f2bf(e1[i]); }
    *(u16x8*)(pbase + lh * 8)      = p0;
    *(u16x8*)(pbase + 16 + lh * 8) = p1;
    // DS unit is in-order within a wave; wave-private tile, no barrier needed.

    // ---- reload P^T as B-operand: col q = ll, k interleaved like the A operand ----
    const bf16x16 bp = mk2(pbase + lh * 8, pbase + 16 + lh * 8);

    // ---- O += V(64 x 32k) . P^T(32k x 16q): 4 WMMAs from LDS ----
#pragma unroll
    for (int t = 0; t < 4; ++t) {
      const u16* vp = Vs + (size_t)(16 * t + ll) * VPITCH;
      bf16x16 av = mk2(vp + lh * 8, vp + 16 + lh * 8);
      oacc[t] = wmma_bf16(av, bp, oacc[t]);
    }

    __syncthreads();  // all waves done with buffer `cur` before it is re-DMA'd
  }

  // ---- epilogue: normalize, residual add, write fp32 ----
  const float inv = 1.0f / l_run;
  const int n = q0 + ll;
#pragma unroll
  for (int t = 0; t < 4; ++t) {
#pragma unroll
    for (int i = 0; i < 8; ++i) {
      const int d = 16 * t + i + 8 * lh;
      const int c = h * CD + d;
      const size_t xi = ((size_t)b * CC + c) * CN + n;
      out[xi] = x[xi] + oacc[t][i] * inv;
    }
  }
}

// ---------------------------------------------------------------------------
// Host-side launch
// ---------------------------------------------------------------------------
extern "C" void kernel_launch(void* const* d_in, const int* in_sizes, int n_in,
                              void* d_out, int out_size, void* d_ws, size_t ws_size,
                              hipStream_t stream) {
  const float* x      = (const float*)d_in[0];  // (8,512,32,32)
  const float* proj_v = (const float*)d_in[1];  // (1536,512)
  const float* proj_g = (const float*)d_in[2];  // (1536,)
  const float* proj_b = (const float*)d_in[3];  // (1536,)
  float* out = (float*)d_out;

  // Workspace layout (bf16 buffers, 2 bytes each)
  char* ws = (char*)d_ws;
  u16* wbf = (u16*)(ws);                                       // 1536*512
  u16* xT  = (u16*)(ws + (size_t)CO3 * CC * 2);                // 8*1024*512
  u16* qT  = (u16*)((char*)xT + (size_t)CB * CN * CC * 2);     // 8*8*1024*64
  u16* kT  = (u16*)((char*)qT + (size_t)CB * CH * CN * CD * 2);
  u16* vv  = (u16*)((char*)kT + (size_t)CB * CH * CN * CD * 2);

  k_wnorm<<<CO3, 256, 0, stream>>>(proj_v, proj_g, wbf);

  const int nconv = CB * CN * CC;  // 4,194,304
  k_transpose_x<<<nconv / 256, 256, 0, stream>>>(x, xT);

  k_qkv_gemm<<<dim3(CO3 / 64, CN / 128, CB), 256, 0, stream>>>(wbf, xT, proj_b,
                                                               qT, kT, vv);

  k_attn<<<dim3(CN / 128, CH, CB), 256, 0, stream>>>(qT, kT, vv, x, out);
}
